// MultiHeadSelfAttention_71305047048342
// MI455X (gfx1250) — matrix operand order repaired and checked by reading, not compiled, not run on hardware
//
#include <hip/hip_runtime.h>
#include <hip/hip_bf16.h>
#include <stdint.h>

typedef __bf16 bf16;
typedef __attribute__((ext_vector_type(16))) __bf16 v16bf;
typedef __attribute__((ext_vector_type(8)))  __bf16 v8bf;
typedef __attribute__((ext_vector_type(8)))  float  v8f;

#define BB 4
#define SS 2048
#define EE 1024
#define HH 16
#define HDD 64
#define MM (BB*SS)     // 8192 rows of x
#define N1 (3*EE)      // 3072 qkv columns

static __device__ __forceinline__ bf16 f2bf(float f) {
  unsigned u = __builtin_bit_cast(unsigned, f);
  unsigned r = (u + 0x7FFFu + ((u >> 16) & 1u)) >> 16;   // round-to-nearest-even
  return __builtin_bit_cast(bf16, (unsigned short)r);
}

static __device__ __forceinline__ v8f wmma_bf16(v16bf a, v16bf b, v8f c) {
  return __builtin_amdgcn_wmma_f32_16x16x32_bf16(false, a, false, b, (short)0, c,
                                                 false, false);
}

// A fragment: 16x32 bf16, row-major source with row stride ld (elements).
// Contiguous k order used here (matches load_b_frag): lane holds row m = lane&15; K base = (lane>=16 ? 16 : 0),
// VGPR0-3 = K[base..base+7], VGPR4-7 = K[base+8..base+15].
static __device__ __forceinline__ v16bf load_a_frag(const bf16* p, int ld, int lane) {
  int m  = lane & 15;
  int kb = (lane >> 4) << 4;                 // 0 or 16
  const bf16* row = p + (long)m * ld;
  v8bf lo = *(const v8bf*)(row + kb);
  v8bf hi = *(const v8bf*)(row + kb + 8);
  return __builtin_shufflevector(lo, hi, 0,1,2,3,4,5,6,7,8,9,10,11,12,13,14,15);
}

// B fragment: 32x16 bf16 where storage is "column major" p[n*ld + k].
// Contiguous k order used here (matches load_a_frag): lane holds column n = lane&15; K base = (lane>=16 ? 16 : 0),
// 16 consecutive K values packed pairwise into 8 VGPRs.
static __device__ __forceinline__ v16bf load_b_frag(const bf16* p, int ld, int lane) {
  int n  = lane & 15;
  int kb = (lane >> 4) << 4;                 // 0 or 16
  return *(const v16bf*)(p + (long)n * ld + kb);
}

// Low 32 bits of a generic pointer to LDS == LDS byte offset (ISA 10.2 aperture).
static __device__ __forceinline__ unsigned lds_off(const void* p) {
  return (unsigned)(uintptr_t)p;
}

// Stage a 128-row x 64-elem (128 B/row) bf16 tile into LDS via the async path.
// One thread per row, 8 x b128 per row, tracked on ASYNCcnt.
static __device__ __forceinline__ void async_tile_load(unsigned lds_base,
                                                       const bf16* gbase,
                                                       int tid) {
  unsigned lds = lds_base + (unsigned)tid * 128u;
  uint64_t g = (uint64_t)(uintptr_t)(gbase + (long)tid * EE);
#pragma unroll
  for (int c = 0; c < 8; ++c) {
    asm volatile("global_load_async_to_lds_b128 %0, %1, off"
                 :: "v"(lds + c * 16u), "v"(g + (uint64_t)(c * 16)) : "memory");
  }
}

static __device__ __forceinline__ void wait_async0() {
  asm volatile("s_wait_asynccnt 0" ::: "memory");
}

// ---------------- conversion / transpose kernels ----------------

__global__ void k_cvt_bf16(const float* __restrict__ src, bf16* __restrict__ dst, int n) {
  int i = blockIdx.x * blockDim.x + threadIdx.x;
  if (i < n) dst[i] = f2bf(src[i]);
}

// dst[c*R + r] = src[r*C + c]  (f32 -> bf16, transpose so B-frags are contiguous)
__global__ void k_cvt_transpose(const float* __restrict__ src, bf16* __restrict__ dst,
                                int R, int C) {
  int i = blockIdx.x * blockDim.x + threadIdx.x;
  if (i < R * C) {
    int r = i / C, c = i % C;
    dst[(long)c * R + r] = f2bf(src[i]);
  }
}

// ---------------- shared GEMM core: 128x128 block tile, K-step 64 ----------------
// Block = 4 waves (2x2), each wave owns a 64x64 tile of 4x4 WMMA accumulators.
// A (rows) and B (transposed weights, [n][k]) are double-buffered in LDS via
// GLOBAL_LOAD_ASYNC_TO_LDS_B128; one s_wait_asynccnt + barrier per 32 WMMAs.

#define GEMM_CORE(SRC_A, SRC_B)                                                  \
  __shared__ alignas(32) bf16 Atile[2][128 * 64];                                \
  __shared__ alignas(32) bf16 Btile[2][128 * 64];                                \
  int tid  = threadIdx.x;                                                        \
  int lane = tid & 31;                                                           \
  int wave = tid >> 5;                                                           \
  int wm = (wave >> 1) * 64;                                                     \
  int wn = (wave & 1) * 64;                                                      \
  int m0b = blockIdx.y * 128;                                                    \
  int n0b = blockIdx.x * 128;                                                    \
  v8f acc[4][4];                                                                 \
  v8f zz = {0.f,0.f,0.f,0.f,0.f,0.f,0.f,0.f};                                    \
  _Pragma("unroll") for (int i = 0; i < 4; ++i)                                  \
    _Pragma("unroll") for (int j = 0; j < 4; ++j) acc[i][j] = zz;                \
  async_tile_load(lds_off(&Atile[0][0]), (SRC_A) + (long)m0b * EE, tid);         \
  async_tile_load(lds_off(&Btile[0][0]), (SRC_B) + (long)n0b * EE, tid);         \
  int buf = 0;                                                                   \
  for (int k = 0; k < EE; k += 64, buf ^= 1) {                                   \
    wait_async0();                                                               \
    __syncthreads();                                                             \
    if (k + 64 < EE) {                                                           \
      async_tile_load(lds_off(&Atile[buf ^ 1][0]),                               \
                      (SRC_A) + (long)m0b * EE + k + 64, tid);                   \
      async_tile_load(lds_off(&Btile[buf ^ 1][0]),                               \
                      (SRC_B) + (long)n0b * EE + k + 64, tid);                   \
    }                                                                            \
    const bf16* As = &Atile[buf][0];                                             \
    const bf16* Bs = &Btile[buf][0];                                             \
    _Pragma("unroll") for (int ks = 0; ks < 64; ks += 32) {                      \
      v16bf a[4], b[4];                                                          \
      _Pragma("unroll") for (int i = 0; i < 4; ++i)                              \
        a[i] = load_a_frag(As + (wm + 16 * i) * 64 + ks, 64, lane);              \
      _Pragma("unroll") for (int j = 0; j < 4; ++j)                              \
        b[j] = load_b_frag(Bs + (wn + 16 * j) * 64 + ks, 64, lane);              \
      _Pragma("unroll") for (int i = 0; i < 4; ++i)                              \
        _Pragma("unroll") for (int j = 0; j < 4; ++j)                            \
          acc[i][j] = wmma_bf16(a[i], b[j], acc[i][j]);                          \
    }                                                                            \
  }                                                                              \
  int m0 = m0b + wm;                                                             \
  int n0 = n0b + wn;                                                             \
  int half = (lane >> 4) * 8;                                                    \
  int ncol = lane & 15;

// ---------------- GEMM 1: qkv = x @ Wqkv + b, scatter to Q/K/V^T ----------------

__global__ __launch_bounds__(128) void k_gemm_qkv(
    const bf16* __restrict__ xb, const bf16* __restrict__ wT,
    const float* __restrict__ bias,
    bf16* __restrict__ Qb, bf16* __restrict__ Kb, bf16* __restrict__ VTb) {
  GEMM_CORE(xb, wT)
#pragma unroll
  for (int i = 0; i < 4; ++i) {
#pragma unroll
    for (int j = 0; j < 4; ++j) {
      int nglob = n0 + j * 16 + ncol;
      float bv = bias[nglob];
      int h   = nglob / (3 * HDD);
      int rem = nglob % (3 * HDD);
      int sel = rem / HDD;
      int d   = rem % HDD;
#pragma unroll
      for (int r = 0; r < 8; ++r) {
        int mglob = m0 + i * 16 + r + half;
        int bidx = mglob / SS, s = mglob % SS;
        float v = acc[i][j][r] + bv;
        long bh = (long)bidx * HH + h;
        if (sel == 0)      Qb[(bh * SS + s) * HDD + d]  = f2bf(v * 0.125f); // 1/sqrt(64)
        else if (sel == 1) Kb[(bh * SS + s) * HDD + d]  = f2bf(v);
        else               VTb[(bh * HDD + d) * SS + s] = f2bf(v);
      }
    }
  }
}

// ---------------- GEMM 2: out = attn @ Wout + bout (fp32 out) ----------------

__global__ __launch_bounds__(128) void k_gemm_out(
    const bf16* __restrict__ ab, const bf16* __restrict__ wT,
    const float* __restrict__ bias, float* __restrict__ out) {
  GEMM_CORE(ab, wT)
#pragma unroll
  for (int i = 0; i < 4; ++i) {
#pragma unroll
    for (int j = 0; j < 4; ++j) {
      int nglob = n0 + j * 16 + ncol;
      float bv = bias[nglob];
#pragma unroll
      for (int r = 0; r < 8; ++r) {
        int mglob = m0 + i * 16 + r + half;
        out[(long)mglob * EE + nglob] = acc[i][j][r] + bv;
      }
    }
  }
}

// ---------------- attention: flash-style, one wave per 16 queries ----------------

__global__ __launch_bounds__(32) void k_attn(
    const bf16* __restrict__ Qb, const bf16* __restrict__ Kb,
    const bf16* __restrict__ VTb, bf16* __restrict__ attnb) {
  __shared__ alignas(32) bf16 Plds[16 * 32];

  int lane = threadIdx.x & 31;
  int nblk = SS / 16;                       // 128 query blocks per (b,h)
  int bh = blockIdx.x / nblk;
  int q0 = (blockIdx.x % nblk) * 16;
  int b = bh / HH, h = bh % HH;

  const bf16* Qp = Qb  + ((long)bh * SS + q0) * HDD;
  const bf16* Kp = Kb  + (long)bh * SS * HDD;
  const bf16* Vp = VTb + (long)bh * HDD * SS;

  v16bf qa0 = load_a_frag(Qp,      HDD, lane);   // d = 0..31
  v16bf qa1 = load_a_frag(Qp + 32, HDD, lane);   // d = 32..63

  float rmax[8], rsum[8];
  v8f o[4];
  v8f z = {0.f,0.f,0.f,0.f,0.f,0.f,0.f,0.f};
#pragma unroll
  for (int r = 0; r < 8; ++r) { rmax[r] = -1e30f; rsum[r] = 0.f; }
#pragma unroll
  for (int j = 0; j < 4; ++j) o[j] = z;

  int colbase = lane & 15;
  int rowoff  = (lane >> 4) * 8;

  for (int kc = 0; kc < SS; kc += 32) {
    // scores: two 16x16 tiles over this 32-key chunk, K-dim = HD (2 steps of 32)
    v8f s0 = z, s1 = z;
    s0 = wmma_bf16(qa0, load_b_frag(Kp + (long)kc * HDD,             HDD, lane), s0);
    s0 = wmma_bf16(qa1, load_b_frag(Kp + (long)kc * HDD + 32,        HDD, lane), s0);
    s1 = wmma_bf16(qa0, load_b_frag(Kp + (long)(kc + 16) * HDD,      HDD, lane), s1);
    s1 = wmma_bf16(qa1, load_b_frag(Kp + (long)(kc + 16) * HDD + 32, HDD, lane), s1);

    // online softmax (rows live across 16 lanes of each half-wave)
    float p0[8], p1[8], alpha[8];
#pragma unroll
    for (int r = 0; r < 8; ++r) {
      float mx = fmaxf(s0[r], s1[r]);
#pragma unroll
      for (int off = 1; off < 16; off <<= 1) mx = fmaxf(mx, __shfl_xor(mx, off, 16));
      float mnew = fmaxf(rmax[r], mx);
      alpha[r] = __expf(rmax[r] - mnew);
      rmax[r] = mnew;
      p0[r] = __expf(s0[r] - mnew);
      p1[r] = __expf(s1[r] - mnew);
      float ps = p0[r] + p1[r];
#pragma unroll
      for (int off = 1; off < 16; off <<= 1) ps += __shfl_xor(ps, off, 16);
      rsum[r] = rsum[r] * alpha[r] + ps;
    }
#pragma unroll
    for (int j = 0; j < 4; ++j)
#pragma unroll
      for (int r = 0; r < 8; ++r) o[j][r] *= alpha[r];

    // C-layout -> A-layout through LDS (bf16 P tile, 16x32)
#pragma unroll
    for (int r = 0; r < 8; ++r) {
      Plds[(r + rowoff) * 32 + colbase]      = f2bf(p0[r]);
      Plds[(r + rowoff) * 32 + 16 + colbase] = f2bf(p1[r]);
    }
    __syncthreads();
    v16bf pa = load_a_frag(Plds, 32, lane);
#pragma unroll
    for (int j = 0; j < 4; ++j) {
      v16bf vb = load_b_frag(Vp + (long)(j * 16) * SS + kc, SS, lane);
      o[j] = wmma_bf16(pa, vb, o[j]);
    }
    __syncthreads();
  }

  // normalize and store bf16 attention output [B,S,E]
#pragma unroll
  for (int j = 0; j < 4; ++j) {
#pragma unroll
    for (int r = 0; r < 8; ++r) {
      int row = q0 + r + rowoff;
      int d = j * 16 + colbase;
      attnb[((long)b * SS + row) * EE + h * HDD + d] = f2bf(o[j][r] / rsum[r]);
    }
  }
}

// ---------------- host launcher ----------------

extern "C" void kernel_launch(void* const* d_in, const int* in_sizes, int n_in,
                              void* d_out, int out_size, void* d_ws, size_t ws_size,
                              hipStream_t stream) {
  const float* x    = (const float*)d_in[0];
  const float* Wqkv = (const float*)d_in[1];
  const float* bqkv = (const float*)d_in[2];
  const float* Wout = (const float*)d_in[3];
  const float* bout = (const float*)d_in[4];
  float* out = (float*)d_out;

  char* ws = (char*)d_ws;
  size_t off = 0;
  bf16* xb    = (bf16*)(ws + off); off += (size_t)MM * EE * 2;     // 16.8 MB
  bf16* wqkvT = (bf16*)(ws + off); off += (size_t)N1 * EE * 2;     //  6.3 MB
  bf16* woutT = (bf16*)(ws + off); off += (size_t)EE * EE * 2;     //  2.1 MB
  bf16* Qb    = (bf16*)(ws + off); off += (size_t)MM * EE * 2;     // 16.8 MB
  bf16* Kb    = (bf16*)(ws + off); off += (size_t)MM * EE * 2;     // 16.8 MB
  bf16* VTb   = (bf16*)(ws + off); off += (size_t)MM * EE * 2;     // 16.8 MB
  bf16* attnb = (bf16*)(ws + off); off += (size_t)MM * EE * 2;     // 16.8 MB

  k_cvt_bf16<<<(MM * EE) / 256, 256, 0, stream>>>(x, xb, MM * EE);
  k_cvt_transpose<<<(EE * N1) / 256, 256, 0, stream>>>(Wqkv, wqkvT, EE, N1);
  k_cvt_transpose<<<(EE * EE) / 256, 256, 0, stream>>>(Wout, woutT, EE, EE);

  dim3 g1(N1 / 128, MM / 128);
  k_gemm_qkv<<<g1, 128, 0, stream>>>(xb, wqkvT, bqkv, Qb, Kb, VTb);

  k_attn<<<BB * HH * (SS / 16), 32, 0, stream>>>(Qb, Kb, VTb, attnb);

  dim3 g2(EE / 128, MM / 128);
  k_gemm_out<<<g2, 128, 0, stream>>>(attnb, woutT, bout, out);
}
